// EnsembleFeatureLoss_OT_ablation_wo_local_10737418240169
// MI455X (gfx1250) — matrix operand order repaired and checked by reading, not compiled, not run on hardware
//
#include <hip/hip_runtime.h>
#include <hip/hip_bf16.h>

#define E_ 4
#define M_ 4096
#define N_ 4096
#define D_ 1024
#define MAX_ITERS_ 100
#define THRESH_ 0.01f
#define NBLK_ 128   // persistent-kernel grid (co-resident on MI455X)

// 1 = stage GEMM tiles with GLOBAL_LOAD_ASYNC_TO_LDS_B128 (ASYNCcnt path);
// 0 = fall back to global_load_b128 -> ds_store_b128 staging.
#define OT_USE_ASYNC_LDS 1

typedef __attribute__((ext_vector_type(16))) _Float16 v16h;
typedef __attribute__((ext_vector_type(8)))  _Float16 v8h;
typedef __attribute__((ext_vector_type(4)))  _Float16 v4h;
typedef __attribute__((ext_vector_type(8)))  float    v8f;

// ---------------------------------------------------------------------------
// Row L2 normalize (fp32 -> f16). One block per row of D_=1024, 256 threads,
// float4 loads, LDS tree reduce.
// ---------------------------------------------------------------------------
__global__ __launch_bounds__(256)
void ot_normalize(const float* __restrict__ src, _Float16* __restrict__ dst) {
  const int row = blockIdx.x;
  const int tid = threadIdx.x;
  __shared__ float red[256];
  __shared__ float sInv;
  const float4 v = ((const float4*)(src + (size_t)row * D_))[tid];
  red[tid] = v.x * v.x + v.y * v.y + v.z * v.z + v.w * v.w;
  __syncthreads();
  for (int s = 128; s > 0; s >>= 1) {
    if (tid < s) red[tid] += red[tid + s];
    __syncthreads();
  }
  if (tid == 0) sInv = 1.f / fmaxf(sqrtf(red[0]), 1e-12f);
  __syncthreads();
  const float inv = sInv;
  v4h o = {(_Float16)(v.x * inv), (_Float16)(v.y * inv),
           (_Float16)(v.z * inv), (_Float16)(v.w * inv)};
  *(v4h*)(dst + (size_t)row * D_ + tid * 4) = o;
}

// ---------------------------------------------------------------------------
// sim[e] = Snorm[e] (MxD) * Tnorm[e]^T (DxN), f16 in, f32 out, WMMA 16x16x32.
// Block tile 128x128, BK=32, 8 waves in 2(M)x4(N) grid -> wave tile 64x32,
// 8 accumulators per wave. Double-buffered LDS with +8 half padding.
// Staging uses async global->LDS copies (ASYNCcnt) when enabled.
// ---------------------------------------------------------------------------
__global__ __launch_bounds__(256)
void ot_gemm_wmma(const _Float16* __restrict__ Sh, const _Float16* __restrict__ Th,
                  float* __restrict__ sim) {
  __shared__ _Float16 As[2][128][40];
  __shared__ _Float16 Bs[2][128][40];
  const int tid   = threadIdx.x;
  const int lane  = tid & 31;
  const int wid   = tid >> 5;
  const int waveM = (wid & 1) * 64;
  const int waveN = (wid >> 1) * 32;
  const int e   = blockIdx.z;
  const int bm0 = blockIdx.y * 128;
  const int bn0 = blockIdx.x * 128;
  const _Float16* Ag = Sh + (size_t)e * M_ * D_ + (size_t)bm0 * D_;
  const _Float16* Bg = Th + (size_t)e * N_ * D_ + (size_t)bn0 * D_;
  float* simE = sim + (size_t)e * M_ * N_;

  const int ldRow = tid >> 2;        // 0..63 (per round)
  const int ldCol = (tid & 3) * 8;   // 0,8,16,24

  const v8f zero = {0.f, 0.f, 0.f, 0.f, 0.f, 0.f, 0.f, 0.f};
  v8f acc[4][2];
#pragma unroll
  for (int mi = 0; mi < 4; ++mi)
#pragma unroll
    for (int ni = 0; ni < 2; ++ni) acc[mi][ni] = zero;

  auto ldTile = [&](int stage, int kt) {
#pragma unroll
    for (int p = 0; p < 2; ++p) {
      const int row = ldRow + p * 64;
      const int kOff = kt * 32 + ldCol;
      const _Float16* ga = Ag + (size_t)row * D_ + kOff;
      const _Float16* gb = Bg + (size_t)row * D_ + kOff;
#if OT_USE_ASYNC_LDS
      // VGLOBAL GLOBAL_LOAD_ASYNC_TO_LDS_B128: VDST = LDS byte offset VGPR,
      // VADDR = 64-bit global address. Tracked by ASYNCcnt.
      const unsigned la = (unsigned)(uintptr_t)&As[stage][row][ldCol];
      const unsigned lb = (unsigned)(uintptr_t)&Bs[stage][row][ldCol];
      asm volatile("global_load_async_to_lds_b128 %0, %1, off"
                   :: "v"(la), "v"(ga) : "memory");
      asm volatile("global_load_async_to_lds_b128 %0, %1, off"
                   :: "v"(lb), "v"(gb) : "memory");
#else
      v8h a = *(const v8h*)ga;
      v8h b = *(const v8h*)gb;
      *(v8h*)&As[stage][row][ldCol] = a;
      *(v8h*)&Bs[stage][row][ldCol] = b;
#endif
    }
  };

  auto waitStage = [&]() {
#if OT_USE_ASYNC_LDS
    asm volatile("s_wait_asynccnt 0x0" ::: "memory");
#endif
  };

  const int rlo = lane & 15;
  const int khiA = (lane >> 4) * 8;   // A: K chunks {khiA..+7} and {16+khiA..+7}
  const int kbB  = (lane >> 4) * 8;   // B: K chunks {kbB..+7} and {16+kbB..+7}
  auto ldA = [&](int stage, int mBase) -> v16h {
    v8h lo = *(const v8h*)&As[stage][mBase + rlo][khiA];
    v8h hi = *(const v8h*)&As[stage][mBase + rlo][16 + khiA];
    return __builtin_shufflevector(lo, hi, 0, 1, 2, 3, 4, 5, 6, 7, 8, 9, 10, 11, 12, 13, 14, 15);
  };
  auto ldB = [&](int stage, int nBase) -> v16h {
    const _Float16* p0 = &Bs[stage][nBase + rlo][kbB];
    v8h lo = *(const v8h*)p0;
    v8h hi = *(const v8h*)(p0 + 16);
    return __builtin_shufflevector(lo, hi, 0, 1, 2, 3, 4, 5, 6, 7, 8, 9, 10, 11, 12, 13, 14, 15);
  };

  ldTile(0, 0);
  waitStage();
  __syncthreads();
  const int KT = D_ / 32;
  for (int kt = 0; kt < KT; ++kt) {
    const int cur = kt & 1;
    if (kt + 1 < KT) ldTile(cur ^ 1, kt + 1);  // async fill of other stage
    v16h bF[2];
#pragma unroll
    for (int ni = 0; ni < 2; ++ni) bF[ni] = ldB(cur, waveN + ni * 16);
#pragma unroll
    for (int mi = 0; mi < 4; ++mi) {
      v16h aF = ldA(cur, waveM + mi * 16);
#pragma unroll
      for (int ni = 0; ni < 2; ++ni)
        acc[mi][ni] = __builtin_amdgcn_wmma_f32_16x16x32_f16(
            false, aF, false, bF[ni], (short)0, acc[mi][ni], false, false);
    }
    waitStage();      // our async writes to (cur^1) landed in LDS
    __syncthreads();  // publish to all waves before next iteration reads it
  }

  // Epilogue: C/D layout -> VGPR j holds row (j + 8*hi), col = lane&15.
#pragma unroll
  for (int mi = 0; mi < 4; ++mi)
#pragma unroll
    for (int ni = 0; ni < 2; ++ni) {
      const int gm0 = bm0 + waveM + mi * 16 + (lane >> 4) * 8;
      const int col = bn0 + waveN + ni * 16 + rlo;
#pragma unroll
      for (int j = 0; j < 8; ++j)
        simE[(size_t)(gm0 + j) * N_ + col] = acc[mi][ni][j];
    }
}

// ---------------------------------------------------------------------------
// Init: r=c=1, errsum/loss/barrier = 0 (re-done every launch; deterministic).
// ---------------------------------------------------------------------------
__global__ void ot_init(float* rbuf, float* cbuf, float* errsum, float* loss,
                        unsigned* bar) {
  const int t = blockIdx.x * 256 + threadIdx.x;
  const int stride = gridDim.x * 256;
  for (int i = t; i < E_ * M_; i += stride) rbuf[i] = 1.f;
  for (int i = t; i < E_ * N_; i += stride) cbuf[i] = 1.f;
  if (t < E_ * 128) errsum[t] = 0.f;
  if (t < E_) loss[t] = 0.f;
  if (t < 2) bar[t] = 0u;
}

// Sense-reversing grid barrier (128 co-resident blocks).
__device__ __forceinline__ void grid_barrier(unsigned* cnt, volatile unsigned* gen) {
  __threadfence();
  __syncthreads();
  if (threadIdx.x == 0) {
    const unsigned old = *gen;
    if (atomicAdd(cnt, 1u) == NBLK_ - 1u) {
      atomicExch(cnt, 0u);
      __threadfence();
      atomicAdd((unsigned*)gen, 1u);
    } else {
      while (*gen == old) { __builtin_amdgcn_s_sleep(1); }
    }
  }
  __syncthreads();
  __threadfence();
}

// ---------------------------------------------------------------------------
// Persistent Sinkhorn + final loss. K = exp(10*(sim-1)) recomputed on the fly.
// Per-ensemble sticky convergence via errsum[e][iter]; converged ensembles
// skip compute but keep hitting the grid barrier; break when all converge.
// ---------------------------------------------------------------------------
__global__ __launch_bounds__(256)
void ot_sinkhorn_persistent(const float* __restrict__ sim, float* rbuf, float* cbuf,
                            float* errsum, float* loss, unsigned* bar) {
  __shared__ float red[256];
  const int tid = threadIdx.x;
  const int blk = blockIdx.x;
  unsigned* cnt = bar;
  volatile unsigned* gen = bar + 1;

  const int rowsPerBlk = (E_ * M_) / NBLK_;  // 128 (chunk stays inside one ensemble)
  const int rowStart = blk * rowsPerBlk;
  const int e_r = rowStart / M_;
  const int e_c = blk >> 5;                  // 32 blocks per ensemble for columns
  const int colStart = (blk & 31) * 128;
  const float uVal = 1.f / (float)M_;
  const float vVal = 1.f / (float)N_;
  const float invM = 1.f / (float)M_;
  const volatile float* ev = errsum;

  for (int it = 0; it < MAX_ITERS_; ++it) {
    bool anyActive = false, actR = false, actC = false;
#pragma unroll
    for (int e = 0; e < E_; ++e) {
      const float err = (it == 0) ? 3.4e38f : ev[e * 128 + (it - 1)] * invM;
      const bool a = (err >= THRESH_);
      anyActive |= a;
      if (e == e_r) actR = a;
      if (e == e_c) actC = a;
    }
    if (!anyActive) break;

    // ----- phase r: r_new = u / (K @ c), accumulate mean|dr| numerator -----
    if (actR) {
      float localErr = 0.f;
      const float4* c4 = (const float4*)(cbuf + e_r * N_);
      for (int rr = 0; rr < rowsPerBlk; ++rr) {
        const size_t g = (size_t)(rowStart + rr);
        const float4* s4 = (const float4*)(sim + g * (size_t)N_);
        float acc = 0.f;
        for (int j = tid; j < N_ / 4; j += 256) {
          const float4 sv = s4[j];
          const float4 cv = c4[j];
          acc += __expf(10.f * (sv.x - 1.f)) * cv.x + __expf(10.f * (sv.y - 1.f)) * cv.y +
                 __expf(10.f * (sv.z - 1.f)) * cv.z + __expf(10.f * (sv.w - 1.f)) * cv.w;
        }
        red[tid] = acc;
        __syncthreads();
        for (int s = 128; s > 0; s >>= 1) {
          if (tid < s) red[tid] += red[tid + s];
          __syncthreads();
        }
        if (tid == 0) {
          const float rn = uVal / red[0];
          localErr += fabsf(rn - rbuf[g]);
          rbuf[g] = rn;
        }
        __syncthreads();
      }
      if (tid == 0) atomicAdd(&errsum[e_r * 128 + it], localErr);
    }
    grid_barrier(cnt, gen);

    // ----- phase c: c = v / (K^T @ r) ; block owns 128 columns -----
    if (actC) {
      const int colL = tid & 127;
      const int half = tid >> 7;
      const float* simEp = sim + (size_t)e_c * M_ * N_;
      const float* rE = rbuf + e_c * M_;
      float partial = 0.f;
      for (int i = half; i < M_; i += 2)
        partial += __expf(10.f * (simEp[(size_t)i * N_ + colStart + colL] - 1.f)) * rE[i];
      red[tid] = partial;
      __syncthreads();
      if (tid < 128) cbuf[e_c * N_ + colStart + tid] = vVal / (red[tid] + red[tid + 128]);
      __syncthreads();
    }
    grid_barrier(cnt, gen);
  }

  // ----- loss: sum_ij r_i * c_j * K_ij * sim_ij per ensemble -----
  {
    float blockLoss = 0.f;
    const float4* c4 = (const float4*)(cbuf + e_r * N_);
    for (int rr = 0; rr < rowsPerBlk; ++rr) {
      const size_t g = (size_t)(rowStart + rr);
      const float4* s4 = (const float4*)(sim + g * (size_t)N_);
      float acc = 0.f;
      for (int j = tid; j < N_ / 4; j += 256) {
        const float4 sv = s4[j];
        const float4 cv = c4[j];
        acc += __expf(10.f * (sv.x - 1.f)) * sv.x * cv.x +
               __expf(10.f * (sv.y - 1.f)) * sv.y * cv.y +
               __expf(10.f * (sv.z - 1.f)) * sv.z * cv.z +
               __expf(10.f * (sv.w - 1.f)) * sv.w * cv.w;
      }
      red[tid] = acc;
      __syncthreads();
      for (int s = 128; s > 0; s >>= 1) {
        if (tid < s) red[tid] += red[tid + s];
        __syncthreads();
      }
      if (tid == 0) blockLoss += rbuf[g] * red[0];
      __syncthreads();
    }
    if (tid == 0) atomicAdd(&loss[e_r], blockLoss);
  }
}

// ---------------------------------------------------------------------------
// Ensemble weighting: w = exp(loss/prev), normalized * E; total = sum(w*loss).
// ---------------------------------------------------------------------------
__global__ void ot_finalize(const float* __restrict__ loss, const float* __restrict__ prev,
                            float* __restrict__ out) {
  if (threadIdx.x == 0) {
    float l[E_], w[E_], s = 0.f;
    for (int e = 0; e < E_; ++e) {
      l[e] = loss[e];
      w[e] = expf(l[e] / (prev[e] + 1e-8f));
      s += w[e];
    }
    float tot = 0.f;
    for (int e = 0; e < E_; ++e) tot += (w[e] / s * (float)E_) * l[e];
    out[0] = tot;
  }
}

extern "C" void kernel_launch(void* const* d_in, const int* in_sizes, int n_in,
                              void* d_out, int out_size, void* d_ws, size_t ws_size,
                              hipStream_t stream) {
  const float* gts   = (const float*)d_in[0];
  const float* feats = (const float*)d_in[1];
  const float* prev  = (const float*)d_in[2];

  _Float16* Sh = (_Float16*)d_ws;
  _Float16* Th = Sh + (size_t)E_ * M_ * D_;
  float* sim   = (float*)(Th + (size_t)E_ * N_ * D_);
  float* rbuf  = sim + (size_t)E_ * M_ * N_;
  float* cbuf  = rbuf + E_ * M_;
  float* errs  = cbuf + E_ * N_;
  float* loss  = errs + E_ * 128;
  unsigned* bar = (unsigned*)(loss + E_);
  const size_t need = (size_t)((char*)(bar + 2) - (char*)d_ws);
  if (ws_size < need) return;  // workspace too small: bail deterministically

  ot_init<<<32, 256, 0, stream>>>(rbuf, cbuf, errs, loss, bar);
  ot_normalize<<<E_ * M_, 256, 0, stream>>>(gts, Sh);
  ot_normalize<<<E_ * N_, 256, 0, stream>>>(feats, Th);
  ot_gemm_wmma<<<dim3(N_ / 128, M_ / 128, E_), 256, 0, stream>>>(Sh, Th, sim);
  ot_sinkhorn_persistent<<<NBLK_, 256, 0, stream>>>(sim, rbuf, cbuf, errs, loss, bar);
  ot_finalize<<<1, 32, 0, stream>>>(loss, prev, (float*)d_out);
}
